// DovePeakPredictor_75385265979999
// MI455X (gfx1250) — matrix operand order repaired and checked by reading, not compiled, not run on hardware
//
#include <hip/hip_runtime.h>
#include <hip/hip_bf16.h>
#include <stdint.h>

// ---------------------------------------------------------------------------
// DovePeakPredictor for MI455X (gfx1250).
// f16 WMMA (v_wmma_f32_16x16x32_f16) for all matmuls, register-blocked so each
// wave amortizes fragment loads over 8 WMMAs; flash attention with fused ALiBi;
// fused qkv-scatter and SwiGLU epilogues; fp32 residual stream.
// ---------------------------------------------------------------------------

constexpr int kB    = 4;
constexpr int kL    = 2048;
constexpr int kNF   = 10;
constexpr int kDED  = 128;
constexpr int kD    = 128;
constexpr int kH    = 4;
constexpr int kNL   = 4;
constexpr int kHid  = 512;    // one half of the 2*HID SwiGLU projection
constexpr int kHead = 32;
constexpr int kCin  = 2 * kNF + kDED;   // 148

typedef __attribute__((ext_vector_type(16))) _Float16 v16h;
typedef __attribute__((ext_vector_type(8)))  float    v8f;

union frag_u { v16h v; uint4 q[2]; };

// A fragment (16x32, f16): lane = m + 16*hiK; half i holds K = 16*(i/8) + 8*hiK + i%8.
__device__ __forceinline__ v16h load_fragA(const _Float16* base, int row0, int ld, int k0) {
    int lane = threadIdx.x & 31;
    int m = lane & 15, hi = lane >> 4;
    const _Float16* p = base + (size_t)(row0 + m) * ld + k0 + hi * 8;
    frag_u u;
    u.q[0] = *(const uint4*)(p);        // K = 8*hi + 0..7
    u.q[1] = *(const uint4*)(p + 16);   // K = 16 + 8*hi + 0..7
    return u.v;
}

// B fragment (32x16, f16) loaded from B^T stored row-major [N][K]:
// lane = n + 16*hiK; half i holds K = 16*(i/8) + 8*hiK + i%8.
__device__ __forceinline__ v16h load_fragB(const _Float16* bt, int row0, int ld, int k0) {
    int lane = threadIdx.x & 31;
    int n = lane & 15, hi = lane >> 4;
    const _Float16* p = bt + (size_t)(row0 + n) * ld + k0 + hi * 8;
    frag_u u;
    u.q[0] = *(const uint4*)(p);
    u.q[1] = *(const uint4*)(p + 16);
    return u.v;
}

__device__ __forceinline__ v8f wmma16(v16h a, v16h b, v8f c) {
    return __builtin_amdgcn_wmma_f32_16x16x32_f16(false, a, false, b, (short)0, c,
                                                  false, false);
}

// ---------------------------------------------------------------------------
// Register-blocked GEMM core: each wave computes a 32x64 tile of
// C[M,N] = A[M,K](f16) * W[N,K]^T(f16); block = 8 waves stacked over M
// (block tile 256x64). 8 WMMAs per K-step for 6 fragment loads.
// ---------------------------------------------------------------------------
#define GEMM_TILE_BODY(KVAL, WROW_EXPR)                                          \
    int wave = threadIdx.x >> 5, lane = threadIdx.x & 31;                        \
    int m0 = blockIdx.y * 256 + wave * 32;                                       \
    int n0 = blockIdx.x * 64;                                                    \
    v8f acc[2][4] = {};                                                          \
    for (int k0 = 0; k0 < (KVAL); k0 += 32) {                                    \
        v16h a0 = load_fragA(A, m0, (KVAL), k0);                                 \
        v16h a1 = load_fragA(A, m0 + 16, (KVAL), k0);                            \
        if (k0 + 32 < (KVAL)) {                                                  \
            __builtin_prefetch(A + (size_t)(m0 + (lane & 15)) * (KVAL) + k0 + 32, 0, 1); \
        }                                                                        \
        _Pragma("unroll")                                                        \
        for (int j = 0; j < 4; j++) {                                            \
            int wrow = (WROW_EXPR);                                              \
            v16h b = load_fragB(W, wrow, (KVAL), k0);                            \
            acc[0][j] = wmma16(a0, b, acc[0][j]);                                \
            acc[1][j] = wmma16(a1, b, acc[1][j]);                                \
        }                                                                        \
    }                                                                            \
    int hi = lane >> 4, ln = lane & 15;

// C += A*W^T + bias into the fp32 residual stream (N = kD = 128).
__global__ void __launch_bounds__(256)
gemm_residual(const _Float16* __restrict__ A, const _Float16* __restrict__ W,
              const float* __restrict__ bias, float* __restrict__ hout,
              int M, int K) {
    GEMM_TILE_BODY(K, n0 + j * 16)
#pragma unroll
    for (int j = 0; j < 4; j++) {
        int col = n0 + j * 16 + ln;
        float bv = bias[col];
#pragma unroll
        for (int t = 0; t < 2; t++)
#pragma unroll
            for (int r = 0; r < 8; r++) {
                int row = m0 + t * 16 + r + 8 * hi;
                hout[(size_t)row * kD + col] += acc[t][j][r] + bv;
            }
    }
}

// QKV projection with fused head-split scatter:
// q,k -> [B,H,L,32] f16; v -> transposed [B,H,32,L] f16 (B^T layout for P@V).
__global__ void __launch_bounds__(256)
gemm_qkv(const _Float16* __restrict__ A, const _Float16* __restrict__ W,
         const float* __restrict__ bias, _Float16* __restrict__ q,
         _Float16* __restrict__ k, _Float16* __restrict__ vT, int M) {
    GEMM_TILE_BODY(kD, n0 + j * 16)
#pragma unroll
    for (int j = 0; j < 4; j++) {
        int col = n0 + j * 16 + ln;          // 0..383
        float bv = bias[col];
        int which = col / kD;                 // 0=q 1=k 2=v
        int c = col % kD;
        int hh = c / kHead, dd = c % kHead;
#pragma unroll
        for (int t = 0; t < 2; t++)
#pragma unroll
            for (int r = 0; r < 8; r++) {
                int row = m0 + t * 16 + r + 8 * hi;     // token index
                int b = row >> 11, l = row & (kL - 1);
                size_t bhI = (size_t)(b * kH + hh);
                _Float16 val = (_Float16)(acc[t][j][r] + bv);
                if (which == 0)      q[(bhI * kL + l) * kHead + dd] = val;
                else if (which == 1) k[(bhI * kL + l) * kHead + dd] = val;
                else                 vT[(bhI * kHead + dd) * kL + l] = val;
            }
    }
}

// W1 projection with fused SwiGLU: wave's 4 column tiles are split between
// x_proj cols [n0,n0+32) and gate cols [512+n0,512+n0+32); epilogue computes
// silu(gate)*x directly -> act[M,512] f16. No 2*HID intermediate buffer.
__global__ void __launch_bounds__(256)
gemm_swiglu(const _Float16* __restrict__ A, const _Float16* __restrict__ W,
            const float* __restrict__ bias, _Float16* __restrict__ act, int M) {
    GEMM_TILE_BODY(kD, (j < 2) ? (n0 + j * 16) : (kHid + n0 + (j - 2) * 16))
#pragma unroll
    for (int jj = 0; jj < 2; jj++) {
        int colx = n0 + jj * 16 + ln;        // x_proj column in [0,512)
        float bx = bias[colx];
        float bg = bias[kHid + colx];
#pragma unroll
        for (int t = 0; t < 2; t++)
#pragma unroll
            for (int r = 0; r < 8; r++) {
                int row = m0 + t * 16 + r + 8 * hi;
                float xp = acc[t][jj][r] + bx;
                float gt = acc[t][2 + jj][r] + bg;
                float sg = gt / (1.0f + __expf(-gt));
                act[(size_t)row * kHid + colx] = (_Float16)(sg * xp);
            }
    }
}

// ---------------------------------------------------------------------------
// Flash attention with fused ALiBi. One wave owns 32 q-rows (two 16-row
// tiles), so each 32-key chunk costs 4 k/v fragment loads for 8 WMMAs.
// q,k: [B,H,L,32] f16; vT: [B,H,32,L] f16; output o16: [B*L,128] f16.
// ---------------------------------------------------------------------------
__global__ void __launch_bounds__(256)
attn_kernel(const _Float16* __restrict__ q, const _Float16* __restrict__ k,
            const _Float16* __restrict__ vT, _Float16* __restrict__ o16) {
    __shared__ _Float16 plds[8][2][16][32];   // per-wave P staging (C- -> A-layout)
    int wave = threadIdx.x >> 5, lane = threadIdx.x & 31;
    int pair = blockIdx.x * 8 + wave;         // 0..1023
    int npq = kL / 32;                        // 64 q-pairs per (b,h)
    int qp = pair % npq;
    int bh = pair / npq;
    int hh = bh % kH;
    int b  = bh / kH;
    const _Float16* qb = q  + (size_t)bh * kL * kHead;
    const _Float16* kb = k  + (size_t)bh * kL * kHead;
    const _Float16* vb = vT + (size_t)bh * kHead * kL;
    float slope = exp2f(-3.0f + 2.0f * (float)hh);   // 2^linspace(-3,3,4)
    const float scale = 0.17677669529663687f;        // 1/sqrt(32)
    int q0 = qp * 32;
    int hi = lane >> 4, ln = lane & 15;

    v16h aq[2];
    aq[0] = load_fragA(qb, q0,      kHead, 0);       // whole head dim = one K
    aq[1] = load_fragA(qb, q0 + 16, kHead, 0);
    v8f acc[2][2] = {};
    float rmax[2][8], rsum[2][8];
#pragma unroll
    for (int t = 0; t < 2; t++)
#pragma unroll
        for (int r = 0; r < 8; r++) { rmax[t][r] = -3.0e38f; rsum[t][r] = 0.0f; }

    for (int c0 = 0; c0 < kL; c0 += 32) {
        v16h kf0 = load_fragB(kb, c0,      kHead, 0);
        v16h kf1 = load_fragB(kb, c0 + 16, kHead, 0);
#pragma unroll
        for (int t = 0; t < 2; t++) {
            v8f s0v = {}; s0v = wmma16(aq[t], kf0, s0v);
            v8f s1v = {}; s1v = wmma16(aq[t], kf1, s1v);
            _Float16* pt = &plds[wave][t][0][0];
#pragma unroll
            for (int r = 0; r < 8; r++) {
                int m  = r + 8 * hi;
                int mG = q0 + t * 16 + m;
                float s0 = s0v[r] * scale - fabsf((float)(mG - (c0 + ln)))      * slope;
                float s1 = s1v[r] * scale - fabsf((float)(mG - (c0 + 16 + ln))) * slope;
                float mx = fmaxf(s0, s1);
#pragma unroll
                for (int off = 1; off < 16; off <<= 1)
                    mx = fmaxf(mx, __shfl_xor(mx, off, 32));
                float mnew = fmaxf(rmax[t][r], mx);
                float p0 = __expf(s0 - mnew), p1 = __expf(s1 - mnew);
                float ps = p0 + p1;
#pragma unroll
                for (int off = 1; off < 16; off <<= 1)
                    ps += __shfl_xor(ps, off, 32);
                float corr = __expf(rmax[t][r] - mnew);
                rsum[t][r] = rsum[t][r] * corr + ps;
                rmax[t][r] = mnew;
                acc[t][0][r] *= corr;
                acc[t][1][r] *= corr;
                pt[m * 32 + ln]      = (_Float16)p0;
                pt[m * 32 + 16 + ln] = (_Float16)p1;
            }
        }
        asm volatile("s_wait_dscnt 0" ::: "memory");   // P stores visible to reload
        v16h vf0 = load_fragB(vb, 0,  kL, c0);          // v dims 0..15
        v16h vf1 = load_fragB(vb, 16, kL, c0);          // v dims 16..31
#pragma unroll
        for (int t = 0; t < 2; t++) {
            v16h ap = load_fragA(&plds[wave][t][0][0], 0, 32, 0);
            acc[t][0] = wmma16(ap, vf0, acc[t][0]);
            acc[t][1] = wmma16(ap, vf1, acc[t][1]);
        }
    }
#pragma unroll
    for (int t = 0; t < 2; t++)
#pragma unroll
        for (int r = 0; r < 8; r++) {
            float inv = 1.0f / rsum[t][r];
            int m = r + 8 * hi;
            size_t row = (size_t)b * kL + q0 + t * 16 + m;
            o16[row * kD + hh * kHead + ln]      = (_Float16)(acc[t][0][r] * inv);
            o16[row * kD + hh * kHead + 16 + ln] = (_Float16)(acc[t][1][r] * inv);
        }
}

// ---------------------------------------------------------------------------
// Elementwise / small kernels
// ---------------------------------------------------------------------------
__global__ void cvt_f16_kernel(const float* __restrict__ in, _Float16* __restrict__ out,
                               int n) {
    int i = blockIdx.x * 256 + threadIdx.x;
    if (i < n) out[i] = (_Float16)in[i];
}

__global__ void __launch_bounds__(128)
embed_kernel(const float* __restrict__ x, const int* __restrict__ dove_ids,
             const float* __restrict__ dove_embed, const float* __restrict__ W_in,
             const float* __restrict__ b_in, const float* __restrict__ g_in,
             float* __restrict__ h) {
    int t = blockIdx.x;          // token index in [0, B*L)
    int b = t / kL;
    int d = threadIdx.x;
    __shared__ float inbuf[kCin];
    for (int j = d; j < kCin; j += 128) {
        float val;
        if (j < kNF)          { float xv = x[(size_t)t * kNF + j];         val = (xv != xv) ? 0.f : xv;  }
        else if (j < 2 * kNF) { float xv = x[(size_t)t * kNF + (j - kNF)]; val = (xv != xv) ? 0.f : 1.f; }
        else                  { val = dove_embed[(size_t)dove_ids[b] * kDED + (j - 2 * kNF)]; }
        inbuf[j] = val;
    }
    __syncthreads();
    float acc = b_in[d];
    const float* wrow = W_in + (size_t)d * kCin;
    for (int j = 0; j < kCin; j++) acc += wrow[j] * inbuf[j];
    float ss = acc * acc;
#pragma unroll
    for (int off = 16; off; off >>= 1) ss += __shfl_xor(ss, off, 32);
    __shared__ float red[4];
    if ((d & 31) == 0) red[d >> 5] = ss;
    __syncthreads();
    float tot = red[0] + red[1] + red[2] + red[3];
    float r = rsqrtf(tot * (1.0f / kD) + 1e-8f);
    h[(size_t)t * kD + d] = g_in[d] * acc * r;
}

__global__ void __launch_bounds__(128)
rmsnorm_f16_kernel(const float* __restrict__ h, const float* __restrict__ g,
                   _Float16* __restrict__ out) {
    int t = blockIdx.x, d = threadIdx.x;
    float v = h[(size_t)t * kD + d];
    float ss = v * v;
#pragma unroll
    for (int off = 16; off; off >>= 1) ss += __shfl_xor(ss, off, 32);
    __shared__ float red[4];
    if ((d & 31) == 0) red[d >> 5] = ss;
    __syncthreads();
    float tot = red[0] + red[1] + red[2] + red[3];
    float r = rsqrtf(tot * (1.0f / kD) + 1e-8f);
    out[(size_t)t * kD + d] = (_Float16)(g[d] * v * r);
}

__global__ void __launch_bounds__(128)
pool_scores_kernel(const float* __restrict__ h, const float* __restrict__ Wp,
                   const float* __restrict__ bp, float* __restrict__ scores) {
    int t = blockIdx.x, d = threadIdx.x;
    float p = h[(size_t)t * kD + d] * Wp[d];
#pragma unroll
    for (int off = 16; off; off >>= 1) p += __shfl_xor(p, off, 32);
    __shared__ float red[4];
    if ((d & 31) == 0) red[d >> 5] = p;
    __syncthreads();
    if (d == 0) scores[t] = red[0] + red[1] + red[2] + red[3] + bp[0];
}

__global__ void __launch_bounds__(256)
pool_weighted_kernel(const float* __restrict__ h, const float* __restrict__ scores,
                     float* __restrict__ pooled) {
    int b = blockIdx.x, tid = threadIdx.x;
    __shared__ float red[8];
    __shared__ float smax, ssum;
    float mx = -3.0e38f;
    for (int l = tid; l < kL; l += 256) mx = fmaxf(mx, scores[b * kL + l]);
#pragma unroll
    for (int off = 16; off; off >>= 1) mx = fmaxf(mx, __shfl_xor(mx, off, 32));
    if ((tid & 31) == 0) red[tid >> 5] = mx;
    __syncthreads();
    if (tid == 0) {
        float m2 = red[0];
        for (int i = 1; i < 8; i++) m2 = fmaxf(m2, red[i]);
        smax = m2;
    }
    __syncthreads();
    float sm = 0.f;
    for (int l = tid; l < kL; l += 256) sm += __expf(scores[b * kL + l] - smax);
#pragma unroll
    for (int off = 16; off; off >>= 1) sm += __shfl_xor(sm, off, 32);
    if ((tid & 31) == 0) red[tid >> 5] = sm;
    __syncthreads();
    if (tid == 0) {
        float s2 = 0.f;
        for (int i = 0; i < 8; i++) s2 += red[i];
        ssum = s2;
    }
    __syncthreads();
    if (tid < kD) {
        float acc = 0.f;
        for (int l = 0; l < kL; l++)
            acc += __expf(scores[b * kL + l] - smax) * h[((size_t)b * kL + l) * kD + tid];
        pooled[b * kD + tid] = acc / ssum;
    }
}

__global__ void __launch_bounds__(64)
classifier_kernel(const float* __restrict__ pooled, const float* __restrict__ Wc1,
                  const float* __restrict__ bc1, const float* __restrict__ Wc2,
                  const float* __restrict__ bc2, float* __restrict__ out) {
    int b = blockIdx.x, j = threadIdx.x;
    __shared__ float z[64];
    float a = bc1[j];
    for (int d = 0; d < kD; d++) a += Wc1[j * kD + d] * pooled[b * kD + d];
    z[j] = fmaxf(a, 0.f);
    __syncthreads();
    if (j < 7) {
        float o = bc2[j];
        for (int t = 0; t < 64; t++) o += Wc2[j * 64 + t] * z[t];
        out[b * 7 + j] = o;
    }
}

// ---------------------------------------------------------------------------
// Host orchestration
// ---------------------------------------------------------------------------
extern "C" void kernel_launch(void* const* d_in, const int* in_sizes, int n_in,
                              void* d_out, int out_size, void* d_ws, size_t ws_size,
                              hipStream_t stream) {
    const float* x          = (const float*)d_in[0];
    const int*   dove_ids   = (const int*)d_in[1];
    const float* dove_embed = (const float*)d_in[2];
    const float* W_in       = (const float*)d_in[3];
    const float* b_in       = (const float*)d_in[4];
    const float* g_in       = (const float*)d_in[5];
    const float* Wqkv       = (const float*)d_in[6];
    const float* bqkv       = (const float*)d_in[7];
    const float* Wo         = (const float*)d_in[8];
    const float* bo         = (const float*)d_in[9];
    const float* g1         = (const float*)d_in[10];
    const float* g2         = (const float*)d_in[11];
    const float* W1         = (const float*)d_in[12];
    const float* b1         = (const float*)d_in[13];
    const float* W2         = (const float*)d_in[14];
    const float* b2         = (const float*)d_in[15];
    const float* Wp         = (const float*)d_in[16];
    const float* bp         = (const float*)d_in[17];
    const float* Wc1        = (const float*)d_in[18];
    const float* bc1        = (const float*)d_in[19];
    const float* Wc2        = (const float*)d_in[20];
    const float* bc2        = (const float*)d_in[21];

    char* ws = (char*)d_ws;
    size_t off = 0;
    auto alloc = [&](size_t bytes) {
        char* p = ws + off;
        off = (off + bytes + 255) & ~(size_t)255;
        return p;
    };
    _Float16* wqkv16 = (_Float16*)alloc((size_t)kNL * 3 * kD * kD * 2);
    _Float16* wo16   = (_Float16*)alloc((size_t)kNL * kD * kD * 2);
    _Float16* w116   = (_Float16*)alloc((size_t)kNL * 2 * kHid * kD * 2);
    _Float16* w216   = (_Float16*)alloc((size_t)kNL * kD * kHid * 2);
    float*    hbuf   = (float*)   alloc((size_t)kB * kL * kD * 4);
    _Float16* hn16   = (_Float16*)alloc((size_t)kB * kL * kD * 2);
    _Float16* q16    = (_Float16*)alloc((size_t)kB * kH * kL * kHead * 2);
    _Float16* k16    = (_Float16*)alloc((size_t)kB * kH * kL * kHead * 2);
    _Float16* vT16   = (_Float16*)alloc((size_t)kB * kH * kHead * kL * 2);
    _Float16* o16    = (_Float16*)alloc((size_t)kB * kL * kD * 2);
    _Float16* act16  = (_Float16*)alloc((size_t)kB * kL * kHid * 2);
    float*    scores = (float*)   alloc((size_t)kB * kL * 4);
    float*    pooled = (float*)   alloc((size_t)kB * kD * 4);

    const int M = kB * kL;   // 8192 tokens

    // Weight conversion to f16 (tiny; streamed once per launch)
    cvt_f16_kernel<<<(kNL * 3 * kD * kD + 255) / 256, 256, 0, stream>>>(Wqkv, wqkv16, kNL * 3 * kD * kD);
    cvt_f16_kernel<<<(kNL * kD * kD + 255) / 256, 256, 0, stream>>>(Wo, wo16, kNL * kD * kD);
    cvt_f16_kernel<<<(kNL * 2 * kHid * kD + 255) / 256, 256, 0, stream>>>(W1, w116, kNL * 2 * kHid * kD);
    cvt_f16_kernel<<<(kNL * kD * kHid + 255) / 256, 256, 0, stream>>>(W2, w216, kNL * kD * kHid);

    embed_kernel<<<M, 128, 0, stream>>>(x, dove_ids, dove_embed, W_in, b_in, g_in, hbuf);

    for (int i = 0; i < kNL; i++) {
        rmsnorm_f16_kernel<<<M, 128, 0, stream>>>(hbuf, g1 + i * kD, hn16);
        gemm_qkv<<<dim3((3 * kD) / 64, M / 256), 256, 0, stream>>>(
            hn16, wqkv16 + (size_t)i * 3 * kD * kD, bqkv + i * 3 * kD,
            q16, k16, vT16, M);
        attn_kernel<<<(kB * kH * (kL / 32)) / 8, 256, 0, stream>>>(q16, k16, vT16, o16);
        gemm_residual<<<dim3(kD / 64, M / 256), 256, 0, stream>>>(
            o16, wo16 + (size_t)i * kD * kD, bo + i * kD, hbuf, M, kD);
        rmsnorm_f16_kernel<<<M, 128, 0, stream>>>(hbuf, g2 + i * kD, hn16);
        gemm_swiglu<<<dim3(kHid / 32, M / 256), 256, 0, stream>>>(
            hn16, w116 + (size_t)i * 2 * kHid * kD, b1 + i * 2 * kHid, act16, M);
        gemm_residual<<<dim3(kD / 64, M / 256), 256, 0, stream>>>(
            act16, w216 + (size_t)i * kD * kHid, b2 + i * kD, hbuf, M, kHid);
    }

    pool_scores_kernel<<<M, 128, 0, stream>>>(hbuf, Wp, bp, scores);
    pool_weighted_kernel<<<kB, 256, 0, stream>>>(hbuf, scores, pooled);
    classifier_kernel<<<kB, 64, 0, stream>>>(pooled, Wc1, bc1, Wc2, bc2, (float*)d_out);

    (void)in_sizes; (void)n_in; (void)out_size; (void)ws_size;
}
